// Net_86689619902639
// MI455X (gfx1250) — matrix operand order repaired and checked by reading, not compiled, not run on hardware
//
#include <hip/hip_runtime.h>
#include <hip/hip_bf16.h>

typedef __bf16 bf16t;
typedef __attribute__((ext_vector_type(8)))  bf16t v8bf;
typedef __attribute__((ext_vector_type(16))) bf16t v16bf;
typedef __attribute__((ext_vector_type(8)))  float v8f;
typedef __attribute__((ext_vector_type(2)))  float v2f;

#define BN_EPS 1e-5f
#define KT   128   // K-tile staged in LDS (elements); every conv CKKp is a multiple of 128
#define KTP  136   // padded LDS row length (+16B) to spread fragment reads across banks

__device__ __forceinline__ v16bf cat8(v8bf lo, v8bf hi) {
  v16bf r;
#pragma unroll
  for (int i = 0; i < 8; ++i) { r[i] = lo[i]; r[i + 8] = hi[i]; }
  return r;
}

// ---------------- weight prep ----------------
__global__ __launch_bounds__(256) void cvt_pad_w_bf16(const float* __restrict__ src,
                                                      bf16t* __restrict__ dst,
                                                      int OC, int CKK, int OCp, int CKKp, int nmat) {
  long idx = (long)blockIdx.x * blockDim.x + threadIdx.x;
  long total = (long)nmat * OCp * CKKp;
  if (idx >= total) return;
  int kk = (int)(idx % CKKp);
  long t = idx / CKKp;
  int oc = (int)(t % OCp);
  int m  = (int)(t / OCp);
  float v = 0.f;
  if (oc < OC && kk < CKK) v = src[((long)m * OC + oc) * CKK + kk];
  dst[idx] = (bf16t)v;
}

__global__ __launch_bounds__(256) void pad_w_f32(const float* __restrict__ src,
                                                 float* __restrict__ dst,
                                                 int OC, int K, int OCp, int nmat) {
  long idx = (long)blockIdx.x * blockDim.x + threadIdx.x;
  long total = (long)nmat * OCp * K;
  if (idx >= total) return;
  int k = (int)(idx % K);
  long t = idx / K;
  int oc = (int)(t % OCp);
  int m  = (int)(t / OCp);
  dst[idx] = (oc < OC) ? src[((long)m * OC + oc) * K + k] : 0.f;
}

// ---------------- im2col (patch-major bf16, zero padded to CKKp/NPp) ----------------
__global__ __launch_bounds__(256) void im2col_bf16(const float* __restrict__ src,
                                                   bf16t* __restrict__ col,
                                                   int C, int H, int W, int K, int stride, int pad,
                                                   int OH, int OW, int N,
                                                   long strideC, long strideN,
                                                   int CKK, int CKKp, long NP, long NPp) {
  long idx = (long)blockIdx.x * blockDim.x + threadIdx.x;
  long total = NPp * (long)CKKp;
  if (idx >= total) return;
  int kk = (int)(idx % CKKp);
  long np = idx / CKKp;
  float v = 0.f;
  if (np < NP && kk < CKK) {
    int KK = K * K;
    int c = kk / KK; int r = kk % KK; int kh = r / K; int kw = r % K;
    long n = np / ((long)OH * OW); int r2 = (int)(np % ((long)OH * OW));
    int oh = r2 / OW, ow = r2 % OW;
    int ih = oh * stride - pad + kh;
    int iw = ow * stride - pad + kw;
    if (ih >= 0 && ih < H && iw >= 0 && iw < W)
      v = src[n * strideN + (long)c * strideC + (long)ih * W + iw];
  }
  col[idx] = (bf16t)v;
}

// ---------------- conv GEMM: bf16 WMMA + async global->LDS double-buffered B tiles --------
// A: bf16 [OCp][CKKp] row-major (per-wave, direct from global).
// B: bf16 [NPp][CKKp] patch-major. The 64xKT B tile is SHARED by all 4 waves of a block:
// staged once per block into LDS with global_load_async_to_lds_b128 (ASYNCcnt), double-buffered.
// Output: f32 [OC][NPfull] (layout C,N,H,W) with fused bias + inference-BN + ReLU.
__global__ __launch_bounds__(128) void gemm_bf16_wmma(
    const bf16t* __restrict__ A, const bf16t* __restrict__ B, float* __restrict__ out,
    int CKKp, long NPp, long NPfull, int OC, int OCp,
    const float* __restrict__ bias,
    const float* __restrict__ bng, const float* __restrict__ bnb,
    const float* __restrict__ bnm, const float* __restrict__ bnv) {
  __shared__ bf16t lbuf[2][64 * KTP];
  const int tid  = threadIdx.x;
  const int lane = tid & 31;
  const int wave = tid >> 5;
  const int mt   = blockIdx.y * 4 + wave;
  const bool active = (mt * 16 < OCp);
  const int hf   = lane >> 4;
  const int l16  = lane & 15;
  const long nbase = (long)blockIdx.x * 64;
  const bf16t* arow = A + (long)((active ? mt : 0) * 16 + l16) * CKKp;
  const unsigned long long bbase = (unsigned long long)B;
  const int ntiles = CKKp / KT;

  // stage one 64 x KT bf16 tile (16 KB): 128 threads x 8 async b128 loads
  auto stage = [&](int kt, int buf) {
    unsigned lbase = (unsigned)(size_t)(&lbuf[buf][0]);   // flat LDS addr: low 32 bits = LDS offset
#pragma unroll
    for (int i = 0; i < 8; ++i) {
      int cid = i * 128 + tid;          // 0..1023 chunks of 16B
      int row = cid >> 4;               // 0..63
      int cc  = cid & 15;               // 16B chunk within the KT row
      unsigned voff  = (unsigned)(((nbase + row) * (long)CKKp + kt + cc * 8) * 2);
      unsigned laddr = lbase + (unsigned)(row * (KTP * 2) + cc * 16);
      asm volatile("global_load_async_to_lds_b128 %0, %1, %2"
                   :: "v"(laddr), "v"(voff), "s"(bbase) : "memory");
    }
  };

  v8f acc[4] = {};
  stage(0, 0);
  for (int t = 0; t < ntiles; ++t) {
    if (t + 1 < ntiles) {
      stage((t + 1) * KT, (t + 1) & 1);
      asm volatile("s_wait_asynccnt 0x8" ::: "memory");   // tile t done, t+1 in flight
    } else {
      asm volatile("s_wait_asynccnt 0x0" ::: "memory");
    }
    __syncthreads();                                      // tile t visible to all waves
    if (active) {
      const bf16t* lb = &lbuf[t & 1][0];
#pragma unroll
      for (int kk = 0; kk < KT; kk += 32) {
        int k0 = t * KT + kk;
        // A 16x32 bf16 fragment: lane holds row m, K = hf*8+0..7 and 16+hf*8+0..7
        v8bf alo = *(const v8bf*)(arow + k0 + hf * 8);
        v8bf ahi = *(const v8bf*)(arow + k0 + 16 + hf * 8);
        v16bf a = cat8(alo, ahi);
#pragma unroll
        for (int j = 0; j < 4; ++j) {
          // B 32x16 fragment from LDS: lane holds col n, K = hf*8+0..7 and 16+hf*8+0..7
          const bf16t* bp = lb + (j * 16 + l16) * KTP + kk + hf * 8;
          v8bf blo = *(const v8bf*)(bp);
          v8bf bhi = *(const v8bf*)(bp + 16);
          v16bf bb = cat8(blo, bhi);
          acc[j] = __builtin_amdgcn_wmma_f32_16x16x32_bf16(false, a, false, bb, (short)0,
                                                           acc[j], false, false);
        }
      }
    }
    __syncthreads();                                      // all waves done reading buf[t&1]
  }
  if (!active) return;
  // epilogue: D VGPR v -> row mt*16 + hf*8 + v, col = l16
#pragma unroll
  for (int j = 0; j < 4; ++j) {
    long np = nbase + j * 16 + l16;
    if (np >= NPfull) continue;
#pragma unroll
    for (int v = 0; v < 8; ++v) {
      int oc = mt * 16 + hf * 8 + v;
      if (oc >= OC) continue;
      float inv = bng[oc] * rsqrtf(bnv[oc] + BN_EPS);
      float val = (acc[j][v] + bias[oc]) * inv + (bnb[oc] - bnm[oc] * inv);
      out[(long)oc * NPfull + np] = fmaxf(val, 0.f);
    }
  }
}

// ---------------- FC GEMM: f32 WMMA (memory-bound; stream f32 weights directly) ----------
__global__ __launch_bounds__(128) void gemm_f32_wmma(
    const float* __restrict__ A, const float* __restrict__ X, float* __restrict__ outT,
    int K, int OC, int OCp, int outStride,
    const float* __restrict__ bias,
    const float* __restrict__ bng, const float* __restrict__ bnb,
    const float* __restrict__ bnm, const float* __restrict__ bnv,
    int relu) {
  const int lane = threadIdx.x & 31;
  const int wave = threadIdx.x >> 5;
  const int mt = blockIdx.y * 4 + wave;
  if (mt * 16 >= OCp) return;
  const int hf  = lane >> 4;
  const int l16 = lane & 15;
  const float* arow = A + (long)(mt * 16 + l16) * K;
  const float* xrow = X + (long)l16 * K;
  v8f acc[4] = {};   // 4 independent accumulation chains hide WMMA latency
  for (int k0 = 0; k0 < K; k0 += 16) {
    __builtin_prefetch(arow + k0 + 64, 0, 3);
#pragma unroll
    for (int u = 0; u < 4; ++u) {
      int k = k0 + u * 4 + hf * 2;   // f32 A/B fragments: lane holds K = hf*2 + {0,1}
      v2f av = *(const v2f*)(arow + k);
      v2f bv = *(const v2f*)(xrow + k);
      acc[u] = __builtin_amdgcn_wmma_f32_16x16x4_f32(false, av, false, bv, (short)0,
                                                     acc[u], false, false);
    }
  }
  v8f s = acc[0] + acc[1] + acc[2] + acc[3];
#pragma unroll
  for (int v = 0; v < 8; ++v) {
    int oc = mt * 16 + hf * 8 + v;
    if (oc >= OC) continue;
    float val = s[v] + bias[oc];
    if (bng) {
      float inv = bng[oc] * rsqrtf(bnv[oc] + BN_EPS);
      val = val * inv + (bnb[oc] - bnm[oc] * inv);
    }
    if (relu) val = fmaxf(val, 0.f);
    outT[(long)l16 * outStride + oc] = val;
  }
}

// ---------------- pooling / reshape ----------------
__global__ __launch_bounds__(256) void maxpool3x3s2(const float* __restrict__ in,
                                                    float* __restrict__ out,
                                                    int C, int N, int H, int W, int OH, int OW) {
  long idx = (long)blockIdx.x * blockDim.x + threadIdx.x;
  long total = (long)C * N * OH * OW;
  if (idx >= total) return;
  int ow = (int)(idx % OW); long t = idx / OW;
  int oh = (int)(t % OH); t /= OH;
  int n  = (int)(t % N);  int c = (int)(t / N);
  const float* base = in + ((long)c * N + n) * H * W;
  float m = -3.4e38f;
#pragma unroll
  for (int i = 0; i < 3; ++i)
#pragma unroll
    for (int j = 0; j < 3; ++j)
      m = fmaxf(m, base[(long)(oh * 2 + i) * W + (ow * 2 + j)]);
  out[idx] = m;
}

__global__ __launch_bounds__(256) void gate_adaptive_pool(const float* __restrict__ in,
                                                          float* __restrict__ fcin) {
  int idx = blockIdx.x * blockDim.x + threadIdx.x;
  if (idx >= 16 * 64 * 9) return;
  int j = idx % 3; int t = idx / 3;
  int i = t % 3;   t /= 3;
  int c = t % 64;  int n = t / 64;
  int h0 = (i * 28) / 3, h1 = ((i + 1) * 28 + 2) / 3;
  int w0 = (j * 28) / 3, w1 = ((j + 1) * 28 + 2) / 3;
  const float* base = in + ((long)c * 16 + n) * 784;
  float s = 0.f;
  for (int h = h0; h < h1; ++h)
    for (int w = w0; w < w1; ++w)
      s += base[h * 28 + w];
  fcin[n * 576 + c * 9 + i * 3 + j] = s / (float)((h1 - h0) * (w1 - w0));
}

__global__ __launch_bounds__(256) void flatten_cnhw(const float* __restrict__ in,
                                                    float* __restrict__ fcin) {
  int idx = blockIdx.x * blockDim.x + threadIdx.x;
  if (idx >= 16 * 9216) return;
  int n = idx / 9216; int k = idx % 9216;
  int c = k / 36; int hw = k % 36;
  fcin[idx] = in[((long)c * 16 + n) * 36 + hw];
}

// ---------------- gate head ----------------
__global__ void gate_head(const float* __restrict__ logits,  // [16][16], cols 0..3 valid
                          const float* __restrict__ tptr,
                          float* __restrict__ gates) {       // [16][4]
  int b = threadIdx.x;
  if (b >= 16) return;
  float t = fminf(fmaxf(tptr[0], 0.1f), 3.0f);
  float lg[4];
#pragma unroll
  for (int e = 0; e < 4; ++e) lg[e] = logits[b * 16 + e] / t;
  int i0 = 0; float v0 = lg[0];
#pragma unroll
  for (int e = 1; e < 4; ++e) if (lg[e] > v0) { v0 = lg[e]; i0 = e; }
  int i1 = -1; float v1 = -3.4e38f;
#pragma unroll
  for (int e = 0; e < 4; ++e) if (e != i0 && lg[e] > v1) { v1 = lg[e]; i1 = e; }
  float m = fmaxf(v0, v1);
  float e0 = __expf(v0 - m), e1 = __expf(v1 - m);
  float s = e0 + e1;
  float g0 = e0 / s, g1 = e1 / s;
  float s2 = g0 + g1 + 1e-10f;
  g0 /= s2; g1 /= s2;
#pragma unroll
  for (int e = 0; e < 4; ++e) gates[b * 4 + e] = 0.f;
  gates[b * 4 + i0] = g0;
  gates[b * 4 + i1] = g1;
}

__global__ void combine_moe(const float* __restrict__ gates, const float* __restrict__ eout,
                            float* __restrict__ out) {
  int idx = threadIdx.x;
  if (idx >= 160) return;
  int b = idx / 10, o = idx % 10;
  float s = 0.f;
#pragma unroll
  for (int e = 0; e < 4; ++e) s += gates[b * 4 + e] * eout[((long)e * 16 + b) * 16 + o];
  out[idx] = s;
}

// ================= host orchestration =================
extern "C" void kernel_launch(void* const* d_in, const int* in_sizes, int n_in,
                              void* d_out, int out_size, void* d_ws, size_t ws_size,
                              hipStream_t stream) {
  (void)in_sizes; (void)n_in; (void)out_size; (void)ws_size;
  const float* x      = (const float*)d_in[0];
  const float* g_w1   = (const float*)d_in[1];
  const float* g_b1   = (const float*)d_in[2];
  const float* g_w2   = (const float*)d_in[3];
  const float* g_b2   = (const float*)d_in[4];
  const float* g_fc1w = (const float*)d_in[5];
  const float* g_fc1b = (const float*)d_in[6];
  const float* g_fc2w = (const float*)d_in[7];
  const float* g_fc2b = (const float*)d_in[8];
  const float* g_temp = (const float*)d_in[9];
  const float* g_bn1[4] = {(const float*)d_in[10], (const float*)d_in[11], (const float*)d_in[12], (const float*)d_in[13]};
  const float* g_bn2[4] = {(const float*)d_in[14], (const float*)d_in[15], (const float*)d_in[16], (const float*)d_in[17]};
  const float* g_bnf[4] = {(const float*)d_in[18], (const float*)d_in[19], (const float*)d_in[20], (const float*)d_in[21]};
  const float* e_cw[5]  = {(const float*)d_in[22], (const float*)d_in[28], (const float*)d_in[34], (const float*)d_in[40], (const float*)d_in[46]};
  const float* e_cb[5]  = {(const float*)d_in[23], (const float*)d_in[29], (const float*)d_in[35], (const float*)d_in[41], (const float*)d_in[47]};
  const float* e_bng[5] = {(const float*)d_in[24], (const float*)d_in[30], (const float*)d_in[36], (const float*)d_in[42], (const float*)d_in[48]};
  const float* e_bnb[5] = {(const float*)d_in[25], (const float*)d_in[31], (const float*)d_in[37], (const float*)d_in[43], (const float*)d_in[49]};
  const float* e_bnm[5] = {(const float*)d_in[26], (const float*)d_in[32], (const float*)d_in[38], (const float*)d_in[44], (const float*)d_in[50]};
  const float* e_bnv[5] = {(const float*)d_in[27], (const float*)d_in[33], (const float*)d_in[39], (const float*)d_in[45], (const float*)d_in[51]};
  const float* e_fc1w = (const float*)d_in[52];
  const float* e_fc1b = (const float*)d_in[53];
  const float* e_fc2w = (const float*)d_in[54];
  const float* e_fc2b = (const float*)d_in[55];
  const float* e_bnl1[4] = {(const float*)d_in[56], (const float*)d_in[57], (const float*)d_in[58], (const float*)d_in[59]};
  const float* e_bnl2[4] = {(const float*)d_in[60], (const float*)d_in[61], (const float*)d_in[62], (const float*)d_in[63]};
  const float* e_fc3w = (const float*)d_in[64];
  const float* e_fc3b = (const float*)d_in[65];
  float* out = (float*)d_out;

  // ---- workspace bump allocator (~275 MB) ----
  char* wsb = (char*)d_ws;
  size_t off = 0;
  auto alloc = [&](size_t bytes) -> void* {
    void* p = wsb + off;
    off = (off + bytes + 255) & ~(size_t)255;
    return p;
  };
  bf16t* gw1   = (bf16t*)alloc((size_t)32 * 128 * 2);
  bf16t* gw2   = (bf16t*)alloc((size_t)64 * 384 * 2);
  bf16t* ew1   = (bf16t*)alloc((size_t)4 * 64 * 384 * 2);
  bf16t* ew2   = (bf16t*)alloc((size_t)4 * 192 * 1664 * 2);
  bf16t* ew3   = (bf16t*)alloc((size_t)4 * 384 * 1792 * 2);
  bf16t* ew4   = (bf16t*)alloc((size_t)4 * 256 * 3456 * 2);
  bf16t* ew5   = (bf16t*)alloc((size_t)4 * 256 * 2304 * 2);
  float* gfc2w = (float*)alloc((size_t)16 * 128 * 4);
  float* efc3w = (float*)alloc((size_t)4 * 16 * 4096 * 4);
  bf16t* colg1 = (bf16t*)alloc((size_t)200704 * 128 * 2);
  float* gact1 = (float*)alloc((size_t)32 * 200704 * 4);
  float* gpool1= (float*)alloc((size_t)32 * 48400 * 4);
  bf16t* colg2 = (bf16t*)alloc((size_t)12544 * 384 * 2);
  float* gact2 = (float*)alloc((size_t)64 * 12544 * 4);
  float* gfcin = (float*)alloc((size_t)16 * 576 * 4);
  float* gh1   = (float*)alloc((size_t)16 * 128 * 4);
  float* logits= (float*)alloc((size_t)16 * 16 * 4);
  float* gates = (float*)alloc((size_t)16 * 4 * 4);
  bf16t* col1  = (bf16t*)alloc((size_t)48448 * 384 * 2);
  float* eact1 = (float*)alloc((size_t)64 * 48400 * 4);
  float* epool1= (float*)alloc((size_t)64 * 11664 * 4);
  bf16t* col2  = (bf16t*)alloc((size_t)11712 * 1664 * 2);
  float* eact2 = (float*)alloc((size_t)192 * 11664 * 4);
  float* epool2= (float*)alloc((size_t)192 * 2704 * 4);
  bf16t* col3  = (bf16t*)alloc((size_t)2752 * 1792 * 2);
  float* eact3 = (float*)alloc((size_t)384 * 2704 * 4);
  bf16t* col4  = (bf16t*)alloc((size_t)2752 * 3456 * 2);
  float* eact4 = (float*)alloc((size_t)256 * 2704 * 4);
  bf16t* col5  = (bf16t*)alloc((size_t)2752 * 2304 * 2);
  float* eact5 = (float*)alloc((size_t)256 * 2704 * 4);
  float* epool5= (float*)alloc((size_t)256 * 576 * 4);
  float* efcin = (float*)alloc((size_t)16 * 9216 * 4);
  float* eh1   = (float*)alloc((size_t)16 * 4096 * 4);
  float* eh2   = (float*)alloc((size_t)16 * 4096 * 4);
  float* eout  = (float*)alloc((size_t)4 * 16 * 16 * 4);

  auto eblk = [](long n) { return dim3((unsigned)((n + 255) / 256)); };
  auto conv_gemm = [&](const bf16t* Aw, const bf16t* Bc, float* Op, int CKKp, long NPp,
                       long NPfull, int OC, const float* bias, const float* g,
                       const float* bbb, const float* mm, const float* vv) {
    dim3 grid((unsigned)(NPp / 64), (unsigned)((OC + 63) / 64));
    gemm_bf16_wmma<<<grid, 128, 0, stream>>>(Aw, Bc, Op, CKKp, NPp, NPfull, OC, OC,
                                             bias, g, bbb, mm, vv);
  };
  auto fc_gemm = [&](const float* Aw, const float* Xp, float* Op, int K, int OC, int OCp,
                     int outStride, const float* bias, const float* g, const float* bbb,
                     const float* mm, const float* vv, int relu) {
    dim3 grid(1, (unsigned)((OCp + 63) / 64));
    gemm_f32_wmma<<<grid, 128, 0, stream>>>(Aw, Xp, Op, K, OC, OCp, outStride,
                                            bias, g, bbb, mm, vv, relu);
  };

  // ================= gate network =================
  cvt_pad_w_bf16<<<eblk(32 * 128), 256, 0, stream>>>(g_w1, gw1, 32, 75, 32, 128, 1);
  cvt_pad_w_bf16<<<eblk(64 * 384), 256, 0, stream>>>(g_w2, gw2, 64, 288, 64, 384, 1);
  // conv1: 3->32, k5 s2 p2, 224 -> 112
  im2col_bf16<<<eblk((long)200704 * 128), 256, 0, stream>>>(x, colg1, 3, 224, 224, 5, 2, 2,
      112, 112, 16, 50176L, 150528L, 75, 128, 200704L, 200704L);
  conv_gemm(gw1, colg1, gact1, 128, 200704, 200704, 32, g_b1, g_bn1[0], g_bn1[1], g_bn1[2], g_bn1[3]);
  maxpool3x3s2<<<eblk((long)32 * 16 * 55 * 55), 256, 0, stream>>>(gact1, gpool1, 32, 16, 112, 112, 55, 55);
  // conv2: 32->64, k3 s2 p1, 55 -> 28
  im2col_bf16<<<eblk((long)12544 * 384), 256, 0, stream>>>(gpool1, colg2, 32, 55, 55, 3, 2, 1,
      28, 28, 16, 48400L, 3025L, 288, 384, 12544L, 12544L);
  conv_gemm(gw2, colg2, gact2, 384, 12544, 12544, 64, g_b2, g_bn2[0], g_bn2[1], g_bn2[2], g_bn2[3]);
  gate_adaptive_pool<<<eblk(16 * 64 * 9), 256, 0, stream>>>(gact2, gfcin);
  fc_gemm(g_fc1w, gfcin, gh1, 576, 128, 128, 128, g_fc1b, g_bnf[0], g_bnf[1], g_bnf[2], g_bnf[3], 1);
  pad_w_f32<<<eblk(16 * 128), 256, 0, stream>>>(g_fc2w, gfc2w, 4, 128, 16, 1);
  fc_gemm(gfc2w, gh1, logits, 128, 4, 16, 16, g_fc2b, nullptr, nullptr, nullptr, nullptr, 0);
  gate_head<<<1, 32, 0, stream>>>(logits, g_temp, gates);

  // ================= experts (dense, sequential, shared scratch) =================
  cvt_pad_w_bf16<<<eblk((long)4 * 64 * 384), 256, 0, stream>>>(e_cw[0], ew1, 64, 363, 64, 384, 4);
  cvt_pad_w_bf16<<<eblk((long)4 * 192 * 1664), 256, 0, stream>>>(e_cw[1], ew2, 192, 1600, 192, 1664, 4);
  cvt_pad_w_bf16<<<eblk((long)4 * 384 * 1792), 256, 0, stream>>>(e_cw[2], ew3, 384, 1728, 384, 1792, 4);
  cvt_pad_w_bf16<<<eblk((long)4 * 256 * 3456), 256, 0, stream>>>(e_cw[3], ew4, 256, 3456, 256, 3456, 4);
  cvt_pad_w_bf16<<<eblk((long)4 * 256 * 2304), 256, 0, stream>>>(e_cw[4], ew5, 256, 2304, 256, 2304, 4);
  pad_w_f32<<<eblk((long)4 * 16 * 4096), 256, 0, stream>>>(e_fc3w, efc3w, 10, 4096, 16, 4);
  // conv1 im2col of x shared by all 4 experts: 3ch k11 s4 p2, 224 -> 55
  im2col_bf16<<<eblk((long)48448 * 384), 256, 0, stream>>>(x, col1, 3, 224, 224, 11, 4, 2,
      55, 55, 16, 50176L, 150528L, 363, 384, 48400L, 48448L);

  for (int e = 0; e < 4; ++e) {
    conv_gemm(ew1 + (size_t)e * 64 * 384, col1, eact1, 384, 48448, 48400, 64,
              e_cb[0] + e * 64, e_bng[0] + e * 64, e_bnb[0] + e * 64, e_bnm[0] + e * 64, e_bnv[0] + e * 64);
    maxpool3x3s2<<<eblk((long)64 * 16 * 27 * 27), 256, 0, stream>>>(eact1, epool1, 64, 16, 55, 55, 27, 27);
    // conv2: 64->192, k5 s1 p2, 27 -> 27
    im2col_bf16<<<eblk((long)11712 * 1664), 256, 0, stream>>>(epool1, col2, 64, 27, 27, 5, 1, 2,
        27, 27, 16, 11664L, 729L, 1600, 1664, 11664L, 11712L);
    conv_gemm(ew2 + (size_t)e * 192 * 1664, col2, eact2, 1664, 11712, 11664, 192,
              e_cb[1] + e * 192, e_bng[1] + e * 192, e_bnb[1] + e * 192, e_bnm[1] + e * 192, e_bnv[1] + e * 192);
    maxpool3x3s2<<<eblk((long)192 * 16 * 13 * 13), 256, 0, stream>>>(eact2, epool2, 192, 16, 27, 27, 13, 13);
    // conv3: 192->384, k3 s1 p1, 13 -> 13
    im2col_bf16<<<eblk((long)2752 * 1792), 256, 0, stream>>>(epool2, col3, 192, 13, 13, 3, 1, 1,
        13, 13, 16, 2704L, 169L, 1728, 1792, 2704L, 2752L);
    conv_gemm(ew3 + (size_t)e * 384 * 1792, col3, eact3, 1792, 2752, 2704, 384,
              e_cb[2] + e * 384, e_bng[2] + e * 384, e_bnb[2] + e * 384, e_bnm[2] + e * 384, e_bnv[2] + e * 384);
    // conv4: 384->256
    im2col_bf16<<<eblk((long)2752 * 3456), 256, 0, stream>>>(eact3, col4, 384, 13, 13, 3, 1, 1,
        13, 13, 16, 2704L, 169L, 3456, 3456, 2704L, 2752L);
    conv_gemm(ew4 + (size_t)e * 256 * 3456, col4, eact4, 3456, 2752, 2704, 256,
              e_cb[3] + e * 256, e_bng[3] + e * 256, e_bnb[3] + e * 256, e_bnm[3] + e * 256, e_bnv[3] + e * 256);
    // conv5: 256->256
    im2col_bf16<<<eblk((long)2752 * 2304), 256, 0, stream>>>(eact4, col5, 256, 13, 13, 3, 1, 1,
        13, 13, 16, 2704L, 169L, 2304, 2304, 2704L, 2752L);
    conv_gemm(ew5 + (size_t)e * 256 * 2304, col5, eact5, 2304, 2752, 2704, 256,
              e_cb[4] + e * 256, e_bng[4] + e * 256, e_bnb[4] + e * 256, e_bnm[4] + e * 256, e_bnv[4] + e * 256);
    maxpool3x3s2<<<eblk((long)256 * 16 * 6 * 6), 256, 0, stream>>>(eact5, epool5, 256, 16, 13, 13, 6, 6);
    flatten_cnhw<<<eblk((long)16 * 9216), 256, 0, stream>>>(epool5, efcin);
    // FC stack (memory-bound: f32 WMMA straight from f32 weights in HBM)
    fc_gemm(e_fc1w + (size_t)e * 4096 * 9216, efcin, eh1, 9216, 4096, 4096, 4096,
            e_fc1b + (size_t)e * 4096, e_bnl1[0] + (size_t)e * 4096, e_bnl1[1] + (size_t)e * 4096,
            e_bnl1[2] + (size_t)e * 4096, e_bnl1[3] + (size_t)e * 4096, 1);
    fc_gemm(e_fc2w + (size_t)e * 4096 * 4096, eh1, eh2, 4096, 4096, 4096, 4096,
            e_fc2b + (size_t)e * 4096, e_bnl2[0] + (size_t)e * 4096, e_bnl2[1] + (size_t)e * 4096,
            e_bnl2[2] + (size_t)e * 4096, e_bnl2[3] + (size_t)e * 4096, 1);
    fc_gemm(efc3w + (size_t)e * 16 * 4096, eh2, eout + (size_t)e * 256, 4096, 10, 16, 16,
            e_fc3b + e * 10, nullptr, nullptr, nullptr, nullptr, 0);
  }

  combine_moe<<<1, 256, 0, stream>>>(gates, eout, out);
}
